// Model_81930796138745
// MI455X (gfx1250) — matrix operand order repaired and checked by reading, not compiled, not run on hardware
//
#include <hip/hip_runtime.h>

#define HID 30
#define NB  8192
#define NT  512

typedef __attribute__((ext_vector_type(16))) _Float16 v16h;
typedef __attribute__((ext_vector_type(8)))  float    v8f;
typedef __attribute__((ext_vector_type(8)))  unsigned v8u;

// Native v_tanh_f32 on gfx1250 if the toolchain exposes it; exp2-based fallback.
#if __has_builtin(__builtin_amdgcn_tanhf)
__device__ __forceinline__ float ftanh(float x) {
  return __builtin_amdgcn_tanhf(x);
}
__device__ __forceinline__ float fsig(float x) {
  return __builtin_fmaf(0.5f, __builtin_amdgcn_tanhf(0.5f * x), 0.5f);
}
#else
__device__ __forceinline__ float fsig(float x) {
  return __builtin_amdgcn_rcpf(1.0f + __builtin_amdgcn_exp2f(-1.442695041f * x));
}
__device__ __forceinline__ float ftanh(float x) {
  return 2.0f * __builtin_amdgcn_rcpf(1.0f + __builtin_amdgcn_exp2f(-2.885390082f * x)) - 1.0f;
}
#endif

// v_cvt_pk_rtz_f16_f32: lo -> [15:0], hi -> [31:16]
__device__ __forceinline__ unsigned pk2(float lo, float hi) {
  auto p = __builtin_amdgcn_cvt_pkrtz(lo, hi);   // __fp16 ext_vector_type(2)
  return __builtin_bit_cast(unsigned, p);
}
// swap lane halves (lane ^ 16) via ds_swizzle (xor=0x10, and=0x1f)
__device__ __forceinline__ unsigned swap16(unsigned x) {
  return (unsigned)__builtin_amdgcn_ds_swizzle((int)x, 0x401F);
}

// ---------------------------------------------------------------------------
// Setup kernel 1: build A-matrix tiles (8 tiles of 16x32 f16) in a
// WMMA A-layout, contiguous k (K = 16*hi + 2v + idx), stored so the main kernel loads lane-contiguously:
//   ws[m*512 + lane*16 + v*2 + idx]  (halves)
// A row = padded gate G (i:0-31, f:32-63, g:64-95, o:96-127; units 30,31 zero)
// A col K: 0..29 = W_hh[g][K], 30 = W_ih[g], 31 = b_ih[g]+b_hh[g]
// ---------------------------------------------------------------------------
__global__ void build_a(const float* __restrict__ W_ih, const float* __restrict__ W_hh,
                        const float* __restrict__ b_ih, const float* __restrict__ b_hh,
                        _Float16* __restrict__ wsA) {
  int tid = blockIdx.x * blockDim.x + threadIdx.x;   // (m,l,v,idx) = 8*32*8*2 = 4096
  if (tid >= 4096) return;
  int idx = tid & 1;
  int v   = (tid >> 1) & 7;
  int l   = (tid >> 4) & 31;
  int m   = tid >> 9;
  int hi  = l >> 4;
  // 16-bit A 16x32 layout (contiguous, matches B packing): lanes<16: v0-7 -> K=0..15
  //                        lanes>=16: v0-7 -> K=16..31
  int K = (hi ? 16 : 0) + 2 * v + idx;
  int M = l & 15;
  int G = m * 16 + M;          // padded gate row 0..127
  int blk = G >> 5, unit = G & 31;
  float val = 0.0f;
  if (unit < HID) {
    int orig = blk * HID + unit;         // PyTorch gate order i,f,g,o
    if (K < HID)            val = W_hh[orig * HID + K];
    else if (K == HID)      val = W_ih[orig];
    else                    val = b_ih[orig] + b_hh[orig];   // K == 31
  }
  wsA[m * 512 + l * 16 + v * 2 + idx] = (_Float16)val;
}

// Setup kernel 2: padded W_fc (32 floats)
__global__ void build_misc(const float* __restrict__ W_fc, float* __restrict__ wfc) {
  int t = threadIdx.x;
  if (t < 32) wfc[t] = (t < HID) ? W_fc[t] : 0.0f;
}

// Setup kernel 3: transpose x [B,T] f32 -> [T,B] f16 for coalesced per-step loads
__global__ void build_x(const float* __restrict__ x, _Float16* __restrict__ xT) {
  int tid = blockIdx.x * blockDim.x + threadIdx.x;
  if (tid >= NB * NT) return;
  int t = tid >> 13;          // / 8192
  int b = tid & (NB - 1);
  xT[tid] = (_Float16)x[b * NT + t];
}

// ---------------------------------------------------------------------------
// Main kernel: one wave per 16 batch rows. 512 steps, 8 WMMA per step.
//   B matrix (32x16 f16): rows 0..29 = h units, row 30 = x_t, row 31 = 1.0
//   D tiles [gate, batch]: gates on VGPR rows/tiles, batch on lanes
// ---------------------------------------------------------------------------
__global__ __launch_bounds__(256)
void lstm_kernel(const _Float16* __restrict__ wsA, const float* __restrict__ wfc,
                 const unsigned short* __restrict__ xT, const float* __restrict__ b_fc,
                 float* __restrict__ out) {
  const int lane = threadIdx.x & 31;
  const int wid  = (blockIdx.x * blockDim.x + threadIdx.x) >> 5;
  const int b0   = wid * 16;
  const bool hi  = lane >= 16;

  // Loop-invariant weights in A-layout: 8 tiles x 8 VGPR-pairs
  v16h A[8];
#pragma unroll
  for (int m = 0; m < 8; ++m)
    A[m] = *(const v16h*)(wsA + m * 512 + lane * 16);

  // cell/hidden state in D-layout: element (j,r) = unit 16j + r + 8*hi, batch lane%16
  float c[2][8], h[2][8];
#pragma unroll
  for (int j = 0; j < 2; ++j)
#pragma unroll
    for (int r = 0; r < 8; ++r) { c[j][r] = 0.0f; h[j][r] = 0.0f; }

  // B-layout h (packed f16 pairs), starts at zero
  unsigned breg[8] = {0u, 0u, 0u, 0u, 0u, 0u, 0u, 0u};

  const unsigned short* xp = xT + b0 + (lane & 15);
  const v8f zf = {};

  for (int t = 0; t < NT; ++t) {
    // insert x_t (K=30, low half) and 1.0f16 (K=31, high half) on hi lanes
    unsigned xu = (unsigned)(*xp);
    xp += NB;
    unsigned b7 = hi ? (0x3C000000u | xu) : breg[7];

    v8u bu;
    bu[0] = breg[0]; bu[1] = breg[1]; bu[2] = breg[2]; bu[3] = breg[3];
    bu[4] = breg[4]; bu[5] = breg[5]; bu[6] = breg[6]; bu[7] = b7;
    v16h Bv = __builtin_bit_cast(v16h, bu);

    // gates = [h|x|1] x [W_hh;W_ih;bias]^T  -> 8 independent WMMAs
    v8f D[8];
#pragma unroll
    for (int m = 0; m < 8; ++m)
      D[m] = __builtin_amdgcn_wmma_f32_16x16x32_f16(
          false, A[m], false, Bv, (short)0, zf, false, false);

    // per-lane gate nonlinearities + state update (i:D0/D1 f:D2/D3 g:D4/D5 o:D6/D7)
#pragma unroll
    for (int j = 0; j < 2; ++j)
#pragma unroll
      for (int r = 0; r < 8; ++r) {
        float iv = fsig(D[j][r]);
        float fv = fsig(D[2 + j][r]);
        float gv = ftanh(D[4 + j][r]);
        float ov = fsig(D[6 + j][r]);
        float cc = fv * c[j][r] + iv * gv;
        c[j][r] = cc;
        h[j][r] = ov * ftanh(cc);
      }

    // repack h into B-layout: pack f32 pairs -> f16, swap lane halves for the
    // cross-half unit groups, select per half.
#pragma unroll
    for (int q = 0; q < 4; ++q) {
      unsigned p0  = pk2(h[0][2 * q], h[0][2 * q + 1]);  // units (2q,2q+1)+8*hi
      unsigned p1  = pk2(h[1][2 * q], h[1][2 * q + 1]);  // units 16+(2q,2q+1)+8*hi
      unsigned p0s = swap16(p0);
      unsigned p1s = swap16(p1);
      breg[q]     = hi ? p1s : p0;
      breg[4 + q] = hi ? p1  : p0s;
    }
  }

  // out[b] = h_T . W_fc + b_fc ; lane needs wfc[8*hi + r] and wfc[16 + 8*hi + r]
  const int off = hi ? 8 : 0;
  float acc = 0.0f;
#pragma unroll
  for (int r = 0; r < 8; ++r)
    acc += h[0][r] * wfc[off + r] + h[1][r] * wfc[16 + off + r];
  float other = __builtin_bit_cast(float, swap16(__builtin_bit_cast(unsigned, acc)));
  if (!hi) out[b0 + lane] = acc + other + b_fc[0];
}

// ---------------------------------------------------------------------------
extern "C" void kernel_launch(void* const* d_in, const int* in_sizes, int n_in,
                              void* d_out, int out_size, void* d_ws, size_t ws_size,
                              hipStream_t stream) {
  const float* x    = (const float*)d_in[0];
  const float* W_ih = (const float*)d_in[1];
  const float* W_hh = (const float*)d_in[2];
  const float* b_ih = (const float*)d_in[3];
  const float* b_hh = (const float*)d_in[4];
  const float* W_fc = (const float*)d_in[5];
  const float* b_fc = (const float*)d_in[6];

  char* ws = (char*)d_ws;
  _Float16* wsA = (_Float16*)ws;                 // 8 KB  : A tiles
  float*    wfc = (float*)(ws + 8192);           // 128 B : padded W_fc
  _Float16* xT  = (_Float16*)(ws + 16384);       // 8 MB  : x transposed to [T,B] f16

  build_a<<<16, 256, 0, stream>>>(W_ih, W_hh, b_ih, b_hh, wsA);
  build_misc<<<1, 32, 0, stream>>>(W_fc, wfc);
  build_x<<<(NB * NT + 255) / 256, 256, 0, stream>>>(x, xT);

  // 8192/16 = 512 waves, 8 waves per 256-thread block -> 64 blocks
  lstm_kernel<<<64, 256, 0, stream>>>(wsA, wfc, (const unsigned short*)xT,
                                      b_fc, (float*)d_out);
}
